// Block_438086664321
// MI455X (gfx1250) — compile-verified, matrix operand order checked by reading, not run on hardware
//
#include <hip/hip_runtime.h>
#include <hip/hip_bf16.h>

// ---------------------------------------------------------------------------
// CDNA5 (gfx1250) bf16 WMMA transformer block.
//   - V_WMMA_F32_16X16X32_BF16, wave32, 8 waves / 256-thr block
//   - 128x128 GEMM tiles, 32x64 per wave (8 WMMA / K-step), double-buffered LDS
//   - async global->LDS staging (ASYNCcnt) when the builtin is available
// ---------------------------------------------------------------------------

typedef __bf16 bf16;
typedef __attribute__((ext_vector_type(16))) __bf16 v16bf;
typedef __attribute__((ext_vector_type(8)))  float  v8f;
typedef int v4i __attribute__((vector_size(4 * sizeof(int))));

#define EMB   1024
#define NTOK  4096        // B*T = 4*1024
#define SEQ   1024
#define NHEAD 16
#define HDIM  64
#define FFDIM 4096

#if __has_builtin(__builtin_amdgcn_global_load_async_to_lds_b128) && \
    __has_builtin(__builtin_amdgcn_s_wait_asynccnt)
#define USE_ASYNC_LDS 1
#else
#define USE_ASYNC_LDS 0
#endif

static __device__ inline v8f wmma_bf16(v16bf a, v16bf b, v8f c) {
    return __builtin_amdgcn_wmma_f32_16x16x32_bf16(
        /*neg_a=*/false, a, /*neg_b=*/false, b,
        /*c_mod=*/(short)0, c, /*reuse_a=*/false, /*reuse_b=*/false);
}

// Copy 16B (8 bf16) global -> LDS. Async (ASYNCcnt) when available.
static __device__ inline void stage16(const bf16* g, bf16* l) {
#if USE_ASYNC_LDS
    __builtin_amdgcn_global_load_async_to_lds_b128(
        (__attribute__((address_space(1))) v4i*)(void*)g,
        (__attribute__((address_space(3))) v4i*)(void*)l, 0, 0);
#else
    *(uint4*)l = *(const uint4*)g;
#endif
}

static __device__ inline void stage_wait() {
#if USE_ASYNC_LDS
    __builtin_amdgcn_s_wait_asynccnt(0);
#endif
}

// Load a 16x32 bf16 A-layout fragment from a K-contiguous tile.
// ISA layout: lane L holds row M=L%16; K = 8*(L/16)+{0..7} and 8*(L/16)+16+{0..7}.
static __device__ inline v16bf frag_ld(const bf16* tile, int stride) {
    const int lane = threadIdx.x & 31;
    const int row  = lane & 15;
    const int kd   = (lane >> 4) << 2;   // dword offset: 0 or 4 (k=0 or 8)
    const unsigned* p = reinterpret_cast<const unsigned*>(tile + row * stride);
    union { v16bf v; unsigned u[8]; } f;
#pragma unroll
    for (int j = 0; j < 4; ++j) {
        f.u[j]     = p[kd + j];          // k = kbase + 2j, 2j+1
        f.u[4 + j] = p[kd + 8 + j];      // k = kbase + 16 + 2j, ...
    }
    return f.v;
}

// ---------------------------------------------------------------------------
// LayerNorm: one row (1024) per block, 256 threads. Optional f32 / bf16 outputs.
// ---------------------------------------------------------------------------
__global__ __launch_bounds__(256) void layernorm_k(
    const float* __restrict__ in, const float* __restrict__ g,
    const float* __restrict__ be, float* __restrict__ outF,
    bf16* __restrict__ outB)
{
    const int row = blockIdx.x;
    const float* xr = in + (size_t)row * EMB;
    float vals[4];
    float s = 0.f, ss = 0.f;
#pragma unroll
    for (int i = 0; i < 4; ++i) {
        float v = xr[threadIdx.x + i * 256];
        vals[i] = v; s += v; ss += v * v;
    }
#pragma unroll
    for (int m = 1; m < 32; m <<= 1) {
        s  += __shfl_xor(s,  m, 32);
        ss += __shfl_xor(ss, m, 32);
    }
    __shared__ float rs[8], rss[8];
    const int lane = threadIdx.x & 31, wave = threadIdx.x >> 5;
    if (lane == 0) { rs[wave] = s; rss[wave] = ss; }
    __syncthreads();
    if (wave == 0) {
        float a  = (lane < 8) ? rs[lane]  : 0.f;
        float b2 = (lane < 8) ? rss[lane] : 0.f;
#pragma unroll
        for (int m = 1; m < 8; m <<= 1) {
            a  += __shfl_xor(a,  m, 32);
            b2 += __shfl_xor(b2, m, 32);
        }
        if (lane == 0) { rs[0] = a; rss[0] = b2; }
    }
    __syncthreads();
    const float mean = rs[0] * (1.f / EMB);
    const float var  = rss[0] * (1.f / EMB) - mean * mean;
    const float inv  = rsqrtf(var + 1e-5f);
#pragma unroll
    for (int i = 0; i < 4; ++i) {
        int c = threadIdx.x + i * 256;
        float y = (vals[i] - mean) * inv * g[c] + be[c];
        if (outF) outF[(size_t)row * EMB + c] = y;
        if (outB) outB[(size_t)row * EMB + c] = (bf16)y;
    }
}

// ---------------------------------------------------------------------------
// Weight conversion: f32 -> bf16, transposed to [N][K] so B-fragments use the
// same K-contiguous loader as A-fragments.
// ---------------------------------------------------------------------------
__global__ __launch_bounds__(256) void conv_qkv_w(const float* __restrict__ w,
                                                  bf16* __restrict__ bt)
{
    // w: [H=16][K=1024][D=64]; bt: [n=h*64+d][k]  (1M elements)
    int o = blockIdx.x * 256 + threadIdx.x;
    int n = o >> 10, c = o & 1023;
    bt[o] = (bf16)w[((size_t)((n >> 6) << 10 | c) << 6) | (n & 63)];
}

__global__ __launch_bounds__(256) void conv_w_t(const float* __restrict__ w,
                                                bf16* __restrict__ bt,
                                                int K, int N)
{
    // w: [K][N] f32 -> bt: [N][K] bf16
    int o = blockIdx.x * 256 + threadIdx.x;
    int n = o / K, k = o - n * K;
    bt[o] = (bf16)w[(size_t)k * N + n];
}

// ---------------------------------------------------------------------------
// bf16 WMMA GEMM:  C[M,N] = A[M,K] * Bt[N,K]^T  (+ epilogue)
//   block = 256 thr (8 waves, 4Mx2N), 128x128 tile, BK=32, double-buffered LDS
//   per wave: 32x64 output = 8 C-frags, 8 WMMA per K-step
//   MODE 0: store bf16 remapped [B,H,T,D]   (QKV projections)
//   MODE 1: store f32 = acc + bias[n] + res (proj / ffn2 residual add)
//   MODE 2: store bf16 relu(acc + bias[n])  (ffn1)
// ---------------------------------------------------------------------------
template <int MODE>
__global__ __launch_bounds__(256) void gemm_bf16(
    const bf16* __restrict__ A, const bf16* __restrict__ Bt,
    int M, int N, int K,
    const float* __restrict__ bias, const float* __restrict__ res,
    float* __restrict__ outF, bf16* __restrict__ outB)
{
    __shared__ __align__(16) bf16 sA[2][128 * 32];
    __shared__ __align__(16) bf16 sB[2][128 * 32];

    const int tid  = threadIdx.x;
    const int lane = tid & 31;
    const int wave = tid >> 5;
    const int wm = wave & 3;           // 4 waves in M (32 rows each)
    const int wn = wave >> 2;          // 2 waves in N (64 cols each)
    const int m0 = blockIdx.y * 128;
    const int n0 = blockIdx.x * 128;

    const v8f zf = {0.f, 0.f, 0.f, 0.f, 0.f, 0.f, 0.f, 0.f};
    v8f acc[2][4];
#pragma unroll
    for (int i = 0; i < 2; ++i)
#pragma unroll
        for (int j = 0; j < 4; ++j) acc[i][j] = zf;

    const int ldr = tid >> 1;          // 0..127
    const int ldk = (tid & 1) << 4;    // 0 or 16

    // prologue: stage K-step 0 into buffer 0
    {
        const bf16* ga = A + (size_t)(m0 + ldr) * K + ldk;
        const bf16* gb = Bt + (size_t)(n0 + ldr) * K + ldk;
        stage16(ga,     sA[0] + ldr * 32 + ldk);
        stage16(ga + 8, sA[0] + ldr * 32 + ldk + 8);
        stage16(gb,     sB[0] + ldr * 32 + ldk);
        stage16(gb + 8, sB[0] + ldr * 32 + ldk + 8);
    }

    const int nk = K >> 5;
    for (int kt = 0; kt < nk; ++kt) {
        const int buf = kt & 1;
        stage_wait();
        __syncthreads();                 // tile[buf] ready; prev compute done

        if (kt + 1 < nk) {               // stage next K-step into other buffer
            const int k0 = (kt + 1) << 5;
            const bf16* ga = A + (size_t)(m0 + ldr) * K + k0 + ldk;
            const bf16* gb = Bt + (size_t)(n0 + ldr) * K + k0 + ldk;
            bf16* la = sA[buf ^ 1] + ldr * 32 + ldk;
            bf16* lb = sB[buf ^ 1] + ldr * 32 + ldk;
            stage16(ga, la);     stage16(ga + 8, la + 8);
            stage16(gb, lb);     stage16(gb + 8, lb + 8);
        }

        const v16bf a0 = frag_ld(sA[buf] + (wm * 32 +  0) * 32, 32);
        const v16bf a1 = frag_ld(sA[buf] + (wm * 32 + 16) * 32, 32);
#pragma unroll
        for (int j = 0; j < 4; ++j) {
            const v16bf b = frag_ld(sB[buf] + (wn * 64 + j * 16) * 32, 32);
            acc[0][j] = wmma_bf16(a0, b, acc[0][j]);
            acc[1][j] = wmma_bf16(a1, b, acc[1][j]);
        }
    }

    // epilogue
    const int rhalf = (lane >> 4) << 3;
    const int cl    = lane & 15;
#pragma unroll
    for (int i = 0; i < 2; ++i) {
        const int rowBase = m0 + wm * 32 + i * 16 + rhalf;
#pragma unroll
        for (int j = 0; j < 4; ++j) {
            const int n = n0 + wn * 64 + j * 16 + cl;
#pragma unroll
            for (int r = 0; r < 8; ++r) {
                const int m = rowBase + r;
                float v = acc[i][j][r];
                if (MODE == 0) {
                    // [B,T] row -> [B,H,T,D] bf16
                    int b = m >> 10, t = m & 1023, h = n >> 6, d = n & 63;
                    outB[(size_t)(((b << 4) + h) << 10 | t) * 64 + d] = (bf16)v;
                } else if (MODE == 1) {
                    size_t idx = (size_t)m * N + n;
                    outF[idx] = v + bias[n] + res[idx];
                } else {
                    float t = v + bias[n];
                    outB[(size_t)m * N + n] = (bf16)(t > 0.f ? t : 0.f);
                }
            }
        }
    }
}

// ---------------------------------------------------------------------------
// Flash attention (non-causal). grid = (B*H, T/128), 8 waves x 16 query rows.
// QK^T and P*V both via bf16 WMMA; online softmax with shfl row reductions.
// Q/K/V: [B,H,T,64] bf16.  O: [B,T,1024] bf16 (heads concatenated).
// ---------------------------------------------------------------------------
__global__ __launch_bounds__(256) void attn_k(
    const bf16* __restrict__ Q, const bf16* __restrict__ Km,
    const bf16* __restrict__ V, bf16* __restrict__ O)
{
    __shared__ __align__(16) bf16 sK[32 * 64];     // [s][d]
    __shared__ __align__(16) bf16 sVt[64 * 32];    // [d][s]
    __shared__ __align__(16) bf16 sP[8 * 16 * 32]; // per-wave P tile

    const int bh   = blockIdx.x;
    const int tid  = threadIdx.x;
    const int lane = tid & 31;
    const int wave = tid >> 5;
    const int tBase = blockIdx.y * 128 + wave * 16;

    // Q fragments (held in registers for the whole pass)
    const bf16* qbase = Q + ((size_t)bh * SEQ + tBase) * 64;
    const v16bf aq0 = frag_ld(qbase, 64);       // d in [0,32)
    const v16bf aq1 = frag_ld(qbase + 32, 64);  // d in [32,64)

    const v8f zf = {0.f, 0.f, 0.f, 0.f, 0.f, 0.f, 0.f, 0.f};
    v8f of[4] = {zf, zf, zf, zf};
    float mrow[8], lrow[8];
#pragma unroll
    for (int r = 0; r < 8; ++r) { mrow[r] = -3.0e38f; lrow[r] = 0.f; }
    const float scale = 0.125f;   // 1/sqrt(64)

    const int csr = tid >> 3;           // 0..31 : s row for coop loads
    const int cdd = (tid & 7) << 3;     // 0..56 : d offset

    for (int s0 = 0; s0 < SEQ; s0 += 32) {
        __syncthreads();
        // cooperative K tile load (async) + V tile transposed store (sync)
        stage16(Km + ((size_t)bh * SEQ + s0 + csr) * 64 + cdd,
                sK + csr * 64 + cdd);
        union { uint4 q; bf16 h[8]; } tv;
        tv.q = *(const uint4*)(V + ((size_t)bh * SEQ + s0 + csr) * 64 + cdd);
#pragma unroll
        for (int j = 0; j < 8; ++j) sVt[(cdd + j) * 32 + csr] = tv.h[j];
        stage_wait();
        __syncthreads();

        // S = scale * Q K^T  (16x32 per wave, two 16x16 frags)
        v8f sfr[2];
#pragma unroll
        for (int ns = 0; ns < 2; ++ns) {
            v8f c = zf;
            c = wmma_bf16(aq0, frag_ld(sK + (ns * 16) * 64, 64), c);
            c = wmma_bf16(aq1, frag_ld(sK + (ns * 16) * 64 + 32, 64), c);
            sfr[ns] = c;
        }
        float mt[8], alpha[8];
#pragma unroll
        for (int r = 0; r < 8; ++r) {
            float a = sfr[0][r] * scale, b = sfr[1][r] * scale;
            sfr[0][r] = a; sfr[1][r] = b;
            mt[r] = a > b ? a : b;
        }
        // row max across the 16 lanes sharing each row
#pragma unroll
        for (int m = 1; m < 16; m <<= 1)
#pragma unroll
            for (int r = 0; r < 8; ++r) {
                float o = __shfl_xor(mt[r], m, 32);
                mt[r] = mt[r] > o ? mt[r] : o;
            }
        float rsum[8];
#pragma unroll
        for (int r = 0; r < 8; ++r) {
            float mn = mrow[r] > mt[r] ? mrow[r] : mt[r];
            alpha[r] = __expf(mrow[r] - mn);
            mrow[r]  = mn;
            float p0 = __expf(sfr[0][r] - mn);
            float p1 = __expf(sfr[1][r] - mn);
            sfr[0][r] = p0; sfr[1][r] = p1;
            rsum[r] = p0 + p1;
        }
#pragma unroll
        for (int m = 1; m < 16; m <<= 1)
#pragma unroll
            for (int r = 0; r < 8; ++r) rsum[r] += __shfl_xor(rsum[r], m, 32);
#pragma unroll
        for (int r = 0; r < 8; ++r) lrow[r] = lrow[r] * alpha[r] + rsum[r];
        // rescale O accumulators
#pragma unroll
        for (int f = 0; f < 4; ++f)
#pragma unroll
            for (int r = 0; r < 8; ++r) of[f][r] *= alpha[r];

        // P (C-layout f32) -> LDS bf16 in A-layout tile
        bf16* pw = sP + wave * 16 * 32;
        const int prh = (lane >> 4) << 3;
        const int pc  = lane & 15;
#pragma unroll
        for (int f = 0; f < 2; ++f)
#pragma unroll
            for (int r = 0; r < 8; ++r)
                pw[(r + prh) * 32 + f * 16 + pc] = (bf16)sfr[f][r];
        __threadfence_block();

        const v16bf pa = frag_ld(pw, 32);
#pragma unroll
        for (int f = 0; f < 4; ++f)
            of[f] = wmma_bf16(pa, frag_ld(sVt + (f * 16) * 32, 32), of[f]);
    }

    // normalize and store: O[b][t][h*64+d]
    const int b = bh >> 4, h = bh & 15;
    const int trow = tBase + ((lane >> 4) << 3);
#pragma unroll
    for (int f = 0; f < 4; ++f) {
        const int d = h * 64 + f * 16 + (lane & 15);
#pragma unroll
        for (int r = 0; r < 8; ++r) {
            const int t = trow + r;
            O[(size_t)(b * SEQ + t) * EMB + d] = (bf16)(of[f][r] / lrow[r]);
        }
    }
}

// ---------------------------------------------------------------------------
// Host orchestration
// ---------------------------------------------------------------------------
extern "C" void kernel_launch(void* const* d_in, const int* in_sizes, int n_in,
                              void* d_out, int out_size, void* d_ws, size_t ws_size,
                              hipStream_t stream)
{
    (void)in_sizes; (void)n_in; (void)out_size; (void)ws_size;
    const float* x      = (const float*)d_in[0];
    const float* wq     = (const float*)d_in[1];
    const float* wk     = (const float*)d_in[2];
    const float* wv     = (const float*)d_in[3];
    const float* w_proj = (const float*)d_in[4];
    const float* b_proj = (const float*)d_in[5];
    const float* w1     = (const float*)d_in[6];
    const float* b1     = (const float*)d_in[7];
    const float* w2     = (const float*)d_in[8];
    const float* b2     = (const float*)d_in[9];
    const float* g1     = (const float*)d_in[10];
    const float* be1    = (const float*)d_in[11];
    const float* g2     = (const float*)d_in[12];
    const float* be2    = (const float*)d_in[13];
    float* out = (float*)d_out;

    char* wsb = (char*)d_ws;
    size_t off = 0;
    auto alloc = [&](size_t bytes) -> void* {
        void* p = wsb + off;
        off = (off + bytes + 255) & ~(size_t)255;
        return p;
    };
    bf16* wq_t  = (bf16*)alloc((size_t)EMB * EMB * 2);
    bf16* wk_t  = (bf16*)alloc((size_t)EMB * EMB * 2);
    bf16* wv_t  = (bf16*)alloc((size_t)EMB * EMB * 2);
    bf16* wp_t  = (bf16*)alloc((size_t)EMB * EMB * 2);
    bf16* w1_t  = (bf16*)alloc((size_t)EMB * FFDIM * 2);
    bf16* w2_t  = (bf16*)alloc((size_t)FFDIM * EMB * 2);
    bf16* h_b   = (bf16*)alloc((size_t)NTOK * EMB * 2);   // LN1 out; reused as t2
    bf16* q_b   = (bf16*)alloc((size_t)NTOK * EMB * 2);
    bf16* k_b   = (bf16*)alloc((size_t)NTOK * EMB * 2);
    bf16* v_b   = (bf16*)alloc((size_t)NTOK * EMB * 2);
    bf16* o_b   = (bf16*)alloc((size_t)NTOK * EMB * 2);
    float* x1_f = (float*)alloc((size_t)NTOK * EMB * 4);
    float* t1_f = (float*)alloc((size_t)NTOK * EMB * 4);
    bf16* mid_b = (bf16*)alloc((size_t)NTOK * FFDIM * 2);

    // --- weight conversion / transposition to [N][K] bf16 ---
    conv_qkv_w<<<(EMB * EMB) / 256, 256, 0, stream>>>(wq, wq_t);
    conv_qkv_w<<<(EMB * EMB) / 256, 256, 0, stream>>>(wk, wk_t);
    conv_qkv_w<<<(EMB * EMB) / 256, 256, 0, stream>>>(wv, wv_t);
    conv_w_t<<<(EMB * EMB) / 256, 256, 0, stream>>>(w_proj, wp_t, EMB, EMB);
    conv_w_t<<<(EMB * FFDIM) / 256, 256, 0, stream>>>(w1, w1_t, EMB, FFDIM);
    conv_w_t<<<(FFDIM * EMB) / 256, 256, 0, stream>>>(w2, w2_t, FFDIM, EMB);

    // --- LN1 -> h (bf16) ---
    layernorm_k<<<NTOK, 256, 0, stream>>>(x, g1, be1, nullptr, h_b);

    // --- QKV projections (WMMA), remapped to [B,H,T,D] ---
    dim3 gQ(EMB / 128, NTOK / 128);
    gemm_bf16<0><<<gQ, 256, 0, stream>>>(h_b, wq_t, NTOK, EMB, EMB,
                                         nullptr, nullptr, nullptr, q_b);
    gemm_bf16<0><<<gQ, 256, 0, stream>>>(h_b, wk_t, NTOK, EMB, EMB,
                                         nullptr, nullptr, nullptr, k_b);
    gemm_bf16<0><<<gQ, 256, 0, stream>>>(h_b, wv_t, NTOK, EMB, EMB,
                                         nullptr, nullptr, nullptr, v_b);

    // --- flash attention ---
    attn_k<<<dim3(4 * NHEAD, SEQ / 128), 256, 0, stream>>>(q_b, k_b, v_b, o_b);

    // --- output projection + residual -> x1 (f32) ---
    gemm_bf16<1><<<gQ, 256, 0, stream>>>(o_b, wp_t, NTOK, EMB, EMB,
                                         b_proj, x, x1_f, nullptr);

    // --- LN2 applied twice (faithful to reference) ---
    layernorm_k<<<NTOK, 256, 0, stream>>>(x1_f, g2, be2, t1_f, nullptr);
    layernorm_k<<<NTOK, 256, 0, stream>>>(t1_f, g2, be2, nullptr, h_b); // t2 bf16

    // --- FFN ---
    dim3 gF1(FFDIM / 128, NTOK / 128);
    gemm_bf16<2><<<gF1, 256, 0, stream>>>(h_b, w1_t, NTOK, FFDIM, EMB,
                                          b1, nullptr, nullptr, mid_b);
    gemm_bf16<1><<<gQ, 256, 0, stream>>>(mid_b, w2_t, NTOK, EMB, FFDIM,
                                         b2, x1_f, out, nullptr);
}
